// Grapher_11046655886018
// MI455X (gfx1250) — matrix operand order repaired and checked by reading, not compiled, not run on hardware
//
#include <hip/hip_runtime.h>
#include <hip/hip_fp16.h>
#include <math.h>

typedef __attribute__((ext_vector_type(16))) _Float16 v16h;
typedef __attribute__((ext_vector_type(8)))  float    v8f;

#define NB    16
#define NC    64
#define NN    1024
#define BN    (NB * NN)          // 16384
#define KNN   9
#define NHEAD 4
#define NHID  15

#define WMMA_F16(a, b, c) \
  __builtin_amdgcn_wmma_f32_16x16x32_f16(false, (a), false, (b), (short)0, (c), false, false)

// ---------------- fragment helpers (contiguous A/B k order, wave32) ----------------

// A fragment: 16x32 f16, row-major source, leading dim ld (elements).
// lanes 0-15: M=lane, K=0..15; lanes 16-31: M=lane-16, K=16..31
__device__ inline v16h load_a_frag(const _Float16* __restrict__ src, int ld, int lane) {
  int m  = lane & 15;
  int kb = (lane >> 4) * 16;
  const _Float16* r = src + (size_t)m * ld;
  v16h a;
#pragma unroll
  for (int i = 0; i < 16; ++i) { a[i] = r[kb + i]; }
  return a;
}

// B fragment: 32x16 f16. lanes 0-15: N=lane, K=0..15; lanes 16-31: N=lane-16, K=16..31
// B from row-major node matrix: B(k,n) = rows[n*ld + k]   (i.e. h^T)
__device__ inline v16h load_b_from_rows(const _Float16* __restrict__ rows, int ld, int lane) {
  int n  = lane & 15;
  int kb = (lane >> 4) * 16;
  const _Float16* r = rows + (size_t)n * ld + kb;
  v16h b;
#pragma unroll
  for (int i = 0; i < 16; ++i) b[i] = r[i];
  return b;
}

// C/D: lanes 0-15: N=lane, VGPR r -> M=r ; lanes 16-31: N=lane-16, VGPR r -> M=8+r

// ---------------- K0: zero ----------------
__global__ void k_zero(float* __restrict__ p, int n) {
  int t = blockIdx.x * blockDim.x + threadIdx.x;
  if (t < n) p[t] = 0.f;
}

// ---------------- K1: L2 normalize -> h16, sq ----------------
__global__ void k_normalize(const float* __restrict__ x, _Float16* __restrict__ h16,
                            float* __restrict__ sq) {
  int t = blockIdx.x * blockDim.x + threadIdx.x;   // B*N
  if (t >= BN) return;
  int b = t >> 10, n = t & (NN - 1);
  const float* xp = x + (size_t)b * NC * NN + n;
  float s = 0.f;
#pragma unroll 8
  for (int c = 0; c < NC; ++c) { float v = xp[(size_t)c * NN]; s += v * v; }
  float nrm = fmaxf(sqrtf(s), 1e-12f);
  float inv = 1.0f / nrm;
  _Float16* hp = h16 + (size_t)t * NC;
#pragma unroll 8
  for (int c = 0; c < NC; ++c) hp[c] = (_Float16)(xp[(size_t)c * NN] * inv);
  sq[t] = s * inv * inv;
}

// ---------------- K2: WMMA distance + two-level parallel top-9 ----------------
__global__ __launch_bounds__(128) void k_dist_topk(const _Float16* __restrict__ h16,
                                                   const float* __restrict__ sq,
                                                   int* __restrict__ idx) {
  __shared__ float sdist[16 * NN];          // 64 KB; reused for candidate staging
  int blk   = blockIdx.x;                   // B * (N/16)
  int b     = blk / (NN / 16);
  int itile = blk % (NN / 16);
  int tid   = threadIdx.x;
  int wave  = tid >> 5, lane = tid & 31;
  const _Float16* hb = h16 + (size_t)b * NN * NC;

  // A = rows itile*16..+15 of h  (two K-steps of 32)
  const _Float16* arow = hb + (size_t)(itile * 16) * NC;
  v16h a0 = load_a_frag(arow, NC, lane);
  v16h a1 = load_a_frag(arow + 32, NC, lane);

  int nl = lane & 15, mofs = (lane >> 4) * 8;
  float sqi[8];
#pragma unroll
  for (int r = 0; r < 8; ++r) sqi[r] = sq[b * NN + itile * 16 + mofs + r];

  for (int jt = wave; jt < NN / 16; jt += 4) {
    const _Float16* bcols = hb + (size_t)(jt * 16) * NC;
    // prefetch the next j-tile this wave will consume (global_prefetch_b8)
    if (jt + 4 < NN / 16) __builtin_prefetch(bcols + (size_t)4 * 16 * NC, 0, 1);
    v16h b0 = load_b_from_rows(bcols, NC, lane);
    v16h b1 = load_b_from_rows(bcols + 32, NC, lane);
    v8f c = {};
    c = WMMA_F16(a0, b0, c);
    c = WMMA_F16(a1, b1, c);
    float sqj = sq[b * NN + jt * 16 + nl];
#pragma unroll
    for (int r = 0; r < 8; ++r) {
      float d = sqi[r] - 2.f * c[r] + sqj;
      sdist[(mofs + r) * NN + jt * 16 + nl] = d;
    }
  }
  __syncthreads();

  // ---- phase A: 8 threads per row, each scans a 128-column strip (all 128 lanes busy)
  int row = tid >> 3, sub = tid & 7;
  float bv[KNN]; int bi[KNN];
#pragma unroll
  for (int k = 0; k < KNN; ++k) { bv[k] = 3.4e38f; bi[k] = 0; }
  {
    const float* rp = &sdist[row * NN + sub * 128];
    for (int j = 0; j < 128; ++j) {
      float d = rp[j];
      if (d < bv[KNN - 1]) {
        int p = KNN - 1;
        while (p > 0 && bv[p - 1] > d) { bv[p] = bv[p - 1]; bi[p] = bi[p - 1]; --p; }
        bv[p] = d; bi[p] = sub * 128 + j;
      }
    }
  }
  __syncthreads();                          // all strips scanned; dist data now dead

  // ---- stage 16x72 candidates into (reused) LDS
  float* cval = sdist;                      // [16][72]
  int*   cidx = (int*)sdist + 4096;         // [16][72], disjoint region
#pragma unroll
  for (int k = 0; k < KNN; ++k) {
    cval[row * 72 + sub * KNN + k] = bv[k];
    cidx[row * 72 + sub * KNN + k] = bi[k];
  }
  __syncthreads();

  // ---- phase B: 16 threads merge 72 candidates each (strip order preserves tie order)
  if (tid < 16) {
    float mv[KNN]; int mi[KNN];
#pragma unroll
    for (int k = 0; k < KNN; ++k) { mv[k] = 3.4e38f; mi[k] = 0; }
    for (int i = 0; i < 72; ++i) {
      float d = cval[tid * 72 + i];
      if (d < mv[KNN - 1]) {
        int ji = cidx[tid * 72 + i];
        int p = KNN - 1;
        while (p > 0 && mv[p - 1] > d) { mv[p] = mv[p - 1]; mi[p] = mi[p - 1]; --p; }
        mv[p] = d; mi[p] = ji;
      }
    }
    int base = (b * NN + itile * 16 + tid) * KNN;
#pragma unroll
    for (int k = 0; k < KNN; ++k) idx[base + k] = mi[k];
  }
}

// ---------------- K3: Wh per head (WMMA, Fout 15 padded to 16) ----------------
__global__ __launch_bounds__(128) void k_wh_heads(const _Float16* __restrict__ h16,
                                                  const float* __restrict__ W_heads,
                                                  float* __restrict__ Whh) {
  int wid   = blockIdx.x * 4 + (threadIdx.x >> 5);
  int lane  = threadIdx.x & 31;
  int head  = wid & 3;
  int mtile = wid >> 2;                     // BN/16 tiles
  const _Float16* arow = h16 + (size_t)(mtile * 16) * NC;
  v16h a0 = load_a_frag(arow, NC, lane);
  v16h a1 = load_a_frag(arow + 32, NC, lane);

  // B(k,n) = W_heads[head][k][n] for n<15 else 0
  int nB = lane & 15, kbB = (lane >> 4) * 16;
  const float* wp = W_heads + head * NC * NHID;
  v16h b0, b1;
#pragma unroll
  for (int i = 0; i < 16; ++i) {
    int k0 = kbB + i, k1 = 32 + kbB + i;
    b0[i] = (nB < NHID) ? (_Float16)wp[k0 * NHID + nB] : (_Float16)0.f;
    b1[i] = (nB < NHID) ? (_Float16)wp[k1 * NHID + nB] : (_Float16)0.f;
  }
  v8f c = {};
  c = WMMA_F16(a0, b0, c);
  c = WMMA_F16(a1, b1, c);
  int nl = lane & 15, mofs = (lane >> 4) * 8;
#pragma unroll
  for (int r = 0; r < 8; ++r) {
    int mg = mtile * 16 + mofs + r;
    Whh[((size_t)mg * NHEAD + head) * 16 + nl] = c[r];
  }
}

// ---------------- K4: f1/f2 per head ----------------
__global__ void k_f12_heads(const float* __restrict__ Whh, const float* __restrict__ a_heads,
                            float* __restrict__ f1, float* __restrict__ f2) {
  int t = blockIdx.x * blockDim.x + threadIdx.x;   // BN*NHEAD, t = node*4+head
  if (t >= BN * NHEAD) return;
  int head = t & 3;
  const float* wr = Whh + (size_t)t * 16;
  const float* ap = a_heads + head * 2 * NHID;
  float s1 = 0.f, s2 = 0.f;
#pragma unroll
  for (int f = 0; f < NHID; ++f) { s1 += wr[f] * ap[f]; s2 += wr[f] * ap[NHID + f]; }
  f1[t] = s1; f2[t] = s2;
}

// ---------------- K5: edge weights + column denominators (heads) ----------------
__global__ void k_edges_heads(const int* __restrict__ idx, const float* __restrict__ f1,
                              const float* __restrict__ f2, float* __restrict__ wbuf,
                              float* __restrict__ denom) {
  int t = blockIdx.x * blockDim.x + threadIdx.x;   // BN*NHEAD
  if (t >= BN * NHEAD) return;
  int head = t & 3, gn = t >> 2;                   // gn = b*N+i
  int b = gn >> 10;
  const int* ip = idx + (size_t)gn * KNN;
  float f1v = f1[t];
#pragma unroll
  for (int k = 0; k < KNN; ++k) {
    int j = ip[k];
    float e = f1v + f2[((size_t)(b * NN + j)) * NHEAD + head];
    e = (e > 0.f) ? e : 0.2f * e;                  // leaky_relu alpha=0.2
    float w = expf(e);
    wbuf[(size_t)t * KNN + k] = w;
    atomicAdd(&denom[((size_t)(b * NN + j)) * NHEAD + head], w);
  }
}

// ---------------- K6: 9-sparse aggregation + ELU -> hcat (f16, padded to 64) ------
__global__ void k_agg_heads(const int* __restrict__ idx, const float* __restrict__ wbuf,
                            const float* __restrict__ denom, const float* __restrict__ Whh,
                            _Float16* __restrict__ hcat16) {
  int t = blockIdx.x * blockDim.x + threadIdx.x;   // BN*NHEAD
  if (t >= BN * NHEAD) return;
  int head = t & 3, gn = t >> 2;
  int b = gn >> 10;
  const int* ip = idx + (size_t)gn * KNN;
  float acc[NHID];
#pragma unroll
  for (int f = 0; f < NHID; ++f) acc[f] = 0.f;
#pragma unroll
  for (int k = 0; k < KNN; ++k) {
    int j  = ip[k];
    size_t cj = ((size_t)(b * NN + j)) * NHEAD + head;
    float wn = wbuf[(size_t)t * KNN + k] / denom[cj];
    const float* wr = Whh + cj * 16;
#pragma unroll
    for (int f = 0; f < NHID; ++f) acc[f] += wn * wr[f];
  }
  _Float16* out = hcat16 + (size_t)gn * NC + head * NHID;
#pragma unroll
  for (int f = 0; f < NHID; ++f) {
    float v = acc[f];
    v = (v > 0.f) ? v : (expf(v) - 1.f);           // ELU
    out[f] = (_Float16)v;
  }
  if (head == 0) {                                 // zero K-padding 60..63
#pragma unroll
    for (int c = NHEAD * NHID; c < NC; ++c) hcat16[(size_t)gn * NC + c] = (_Float16)0.f;
  }
}

// ---------------- K7: out-layer Wh (WMMA, K 60 padded to 64) ----------------
__global__ __launch_bounds__(128) void k_wh_out(const _Float16* __restrict__ hcat16,
                                                const float* __restrict__ W_out,
                                                float* __restrict__ WhO) {
  int wid   = blockIdx.x * 4 + (threadIdx.x >> 5);
  int lane  = threadIdx.x & 31;
  int ntile = wid & 3;
  int mtile = wid >> 2;
  const _Float16* arow = hcat16 + (size_t)(mtile * 16) * NC;
  v16h a0 = load_a_frag(arow, NC, lane);
  v16h a1 = load_a_frag(arow + 32, NC, lane);

  int nB = lane & 15, kbB = (lane >> 4) * 16;
  int o = ntile * 16 + nB;
  v16h b0, b1;
#pragma unroll
  for (int i = 0; i < 16; ++i) {
    int k0 = kbB + i, k1 = 32 + kbB + i;
    b0[i] = (k0 < NHEAD * NHID) ? (_Float16)W_out[k0 * NC + o] : (_Float16)0.f;
    b1[i] = (k1 < NHEAD * NHID) ? (_Float16)W_out[k1 * NC + o] : (_Float16)0.f;
  }
  v8f c = {};
  c = WMMA_F16(a0, b0, c);
  c = WMMA_F16(a1, b1, c);
  int nl = lane & 15, mofs = (lane >> 4) * 8;
#pragma unroll
  for (int r = 0; r < 8; ++r) {
    int mg = mtile * 16 + mofs + r;
    WhO[(size_t)mg * NC + ntile * 16 + nl] = c[r];
  }
}

// ---------------- K8a: out-layer f1/f2 ----------------
__global__ void k_f12_out(const float* __restrict__ WhO, const float* __restrict__ a_out,
                          float* __restrict__ f1o, float* __restrict__ f2o) {
  int t = blockIdx.x * blockDim.x + threadIdx.x;   // BN
  if (t >= BN) return;
  const float* wr = WhO + (size_t)t * NC;
  float s1 = 0.f, s2 = 0.f;
#pragma unroll 8
  for (int c = 0; c < NC; ++c) { s1 += wr[c] * a_out[c]; s2 += wr[c] * a_out[NC + c]; }
  f1o[t] = s1; f2o[t] = s2;
}

// ---------------- K8b: out-layer edges + denominators ----------------
__global__ void k_edges_out(const int* __restrict__ idx, const float* __restrict__ f1o,
                            const float* __restrict__ f2o, float* __restrict__ wbufO,
                            float* __restrict__ denomO) {
  int t = blockIdx.x * blockDim.x + threadIdx.x;   // BN
  if (t >= BN) return;
  int b = t >> 10;
  const int* ip = idx + (size_t)t * KNN;
  float f1v = f1o[t];
#pragma unroll
  for (int k = 0; k < KNN; ++k) {
    int j = ip[k];
    float e = f1v + f2o[b * NN + j];
    e = (e > 0.f) ? e : 0.2f * e;
    float w = expf(e);
    wbufO[(size_t)t * KNN + k] = w;
    atomicAdd(&denomO[b * NN + j], w);
  }
}

// ---------------- K9: out aggregation + ELU + LeakyReLU(0.01) -> ybuf f16 --------
__global__ void k_agg_out(const int* __restrict__ idx, const float* __restrict__ wbufO,
                          const float* __restrict__ denomO, const float* __restrict__ WhO,
                          _Float16* __restrict__ ybuf16) {
  int t = blockIdx.x * blockDim.x + threadIdx.x;   // BN
  if (t >= BN) return;
  int b = t >> 10;
  const int* ip = idx + (size_t)t * KNN;
  float acc[NC];
#pragma unroll 8
  for (int c = 0; c < NC; ++c) acc[c] = 0.f;
#pragma unroll
  for (int k = 0; k < KNN; ++k) {
    int j = ip[k];
    float wn = wbufO[(size_t)t * KNN + k] / denomO[b * NN + j];
    const float* wr = WhO + (size_t)(b * NN + j) * NC;
#pragma unroll 8
    for (int c = 0; c < NC; ++c) acc[c] += wn * wr[c];
  }
  _Float16* out = ybuf16 + (size_t)t * NC;
#pragma unroll 8
  for (int c = 0; c < NC; ++c) {
    float v = acc[c];
    v = (v > 0.f) ? v : (expf(v) - 1.f);           // ELU
    v = (v > 0.f) ? v : 0.01f * v;                 // LeakyReLU 0.01
    out[c] = (_Float16)v;
  }
}

// ---------------- K10: 1x1 conv (WMMA) + BN + residual ----------------
__global__ __launch_bounds__(128) void k_conv_bn_res(const _Float16* __restrict__ ybuf16,
                                                     const float* __restrict__ conv_w,
                                                     const float* __restrict__ conv_b,
                                                     const float* __restrict__ bn_gamma,
                                                     const float* __restrict__ bn_beta,
                                                     const float* __restrict__ x,
                                                     float* __restrict__ out) {
  int wid   = blockIdx.x * 4 + (threadIdx.x >> 5);
  int lane  = threadIdx.x & 31;
  int otile = wid & 3;
  int mtile = wid >> 2;
  const _Float16* arow = ybuf16 + (size_t)(mtile * 16) * NC;
  v16h a0 = load_a_frag(arow, NC, lane);
  v16h a1 = load_a_frag(arow + 32, NC, lane);

  // B(k=c, n=o) = conv_w[o*64 + c]
  int nB = lane & 15, kbB = (lane >> 4) * 16;
  int o = otile * 16 + nB;
  v16h b0, b1;
#pragma unroll
  for (int i = 0; i < 16; ++i) {
    b0[i] = (_Float16)conv_w[o * NC + kbB + i];
    b1[i] = (_Float16)conv_w[o * NC + 32 + kbB + i];
  }
  v8f c = {};
  c = WMMA_F16(a0, b0, c);
  c = WMMA_F16(a1, b1, c);

  int nl = lane & 15, mofs = (lane >> 4) * 8;
  int oo = otile * 16 + nl;
  float invs  = rsqrtf(1.f + 1e-5f);
  float scale = bn_gamma[oo] * invs;
  float bias  = bn_beta[oo];
  float cb    = conv_b[oo];
#pragma unroll
  for (int r = 0; r < 8; ++r) {
    int mg = mtile * 16 + mofs + r;
    int b  = mg >> 10, nn = mg & (NN - 1);
    size_t xi = (size_t)b * NC * NN + (size_t)oo * NN + nn;
    out[xi] = (c[r] + cb) * scale + bias + x[xi];
  }
}

// ---------------- launcher ----------------
extern "C" void kernel_launch(void* const* d_in, const int* in_sizes, int n_in,
                              void* d_out, int out_size, void* d_ws, size_t ws_size,
                              hipStream_t stream) {
  const float* x        = (const float*)d_in[0];
  const float* W_heads  = (const float*)d_in[1];
  const float* a_heads  = (const float*)d_in[2];
  const float* W_out    = (const float*)d_in[3];
  const float* a_out    = (const float*)d_in[4];
  const float* conv_w   = (const float*)d_in[5];
  const float* conv_b   = (const float*)d_in[6];
  const float* bn_gamma = (const float*)d_in[7];
  const float* bn_beta  = (const float*)d_in[8];
  float* out = (float*)d_out;

  char* p = (char*)d_ws;
  auto take = [&](size_t bytes) { char* q = p; p += (bytes + 255) & ~(size_t)255; return q; };
  _Float16* h16    = (_Float16*)take((size_t)BN * NC * 2);
  float*    sq     = (float*)   take((size_t)BN * 4);
  int*      idx    = (int*)     take((size_t)BN * KNN * 4);
  float*    Whh    = (float*)   take((size_t)BN * NHEAD * 16 * 4);
  float*    f1h    = (float*)   take((size_t)BN * NHEAD * 4);
  float*    f2h    = (float*)   take((size_t)BN * NHEAD * 4);
  float*    wbufH  = (float*)   take((size_t)BN * NHEAD * KNN * 4);
  float*    denomH = (float*)   take((size_t)BN * NHEAD * 4);
  _Float16* hcat16 = (_Float16*)take((size_t)BN * NC * 2);
  float*    WhO    = (float*)   take((size_t)BN * NC * 4);
  float*    f1o    = (float*)   take((size_t)BN * 4);
  float*    f2o    = (float*)   take((size_t)BN * 4);
  float*    wbufO  = (float*)   take((size_t)BN * KNN * 4);
  float*    denomO = (float*)   take((size_t)BN * 4);
  _Float16* ybuf16 = (_Float16*)take((size_t)BN * NC * 2);

  const int T = 256;
  k_zero<<<(BN * NHEAD + T - 1) / T, T, 0, stream>>>(denomH, BN * NHEAD);
  k_zero<<<(BN + T - 1) / T, T, 0, stream>>>(denomO, BN);

  k_normalize<<<(BN + T - 1) / T, T, 0, stream>>>(x, h16, sq);
  k_dist_topk<<<NB * (NN / 16), 128, 0, stream>>>(h16, sq, idx);

  k_wh_heads<<<(BN / 16) * NHEAD / 4, 128, 0, stream>>>(h16, W_heads, Whh);
  k_f12_heads<<<(BN * NHEAD + T - 1) / T, T, 0, stream>>>(Whh, a_heads, f1h, f2h);
  k_edges_heads<<<(BN * NHEAD + T - 1) / T, T, 0, stream>>>(idx, f1h, f2h, wbufH, denomH);
  k_agg_heads<<<(BN * NHEAD + T - 1) / T, T, 0, stream>>>(idx, wbufH, denomH, Whh, hcat16);

  k_wh_out<<<(BN / 16) * 4 / 4, 128, 0, stream>>>(hcat16, W_out, WhO);
  k_f12_out<<<(BN + T - 1) / T, T, 0, stream>>>(WhO, a_out, f1o, f2o);
  k_edges_out<<<(BN + T - 1) / T, T, 0, stream>>>(idx, f1o, f2o, wbufO, denomO);
  k_agg_out<<<(BN + T - 1) / T, T, 0, stream>>>(idx, wbufO, denomO, WhO, ybuf16);

  k_conv_bn_res<<<(BN / 16) * 4 / 4, 128, 0, stream>>>(ybuf16, conv_w, conv_b,
                                                       bn_gamma, bn_beta, x, out);
}
